// LearnedPairConnect_70574902608414
// MI455X (gfx1250) — matrix operand order repaired and checked by reading, not compiled, not run on hardware
//
#include <hip/hip_runtime.h>

typedef _Float16 half8 __attribute__((ext_vector_type(8)));
typedef _Float16 v16h  __attribute__((ext_vector_type(16)));
typedef float    v8f   __attribute__((ext_vector_type(8)));

#define Bc   4
#define Nc   512
#define Dc   48
#define D2c  96
#define HS   104   // row stride (halves) for w1t/w2t and per-wave scratch
#define W3S  72    // row stride (halves) for w3t (K padded to 64)

// Build a v16h A/B fragment from two 16-byte LDS loads.
__device__ __forceinline__ v16h lds2(const _Float16* lo, const _Float16* hi) {
  half8 a = *(const half8*)lo;
  half8 b = *(const half8*)hi;
  return __builtin_shufflevector(a, b, 0,1,2,3,4,5,6,7,8,9,10,11,12,13,14,15);
}

__device__ __forceinline__ v8f wmma_f16(v16h a, v16h b, v8f c) {
  // v_wmma_f32_16x16x32_f16: (neg_a, A, neg_b, B, c_mod, C, reuse_a, reuse_b)
  return __builtin_amdgcn_wmma_f32_16x16x32_f16(false, a, false, b, (short)0, c, false, false);
}

__device__ __forceinline__ v8f splat8(float v) {
  v8f c;
#pragma unroll
  for (int r = 0; r < 8; ++r) c[r] = v;
  return c;
}

// Branch-free erf (Abramowitz-Stegun 7.1.26, |err| <= 1.5e-7): no EXEC
// divergence in the hot loop, uses v_rcp_f32 + v_exp_f32 + 5 FMAs.
__device__ __forceinline__ float erf_fast(float x) {
  const float ax = fabsf(x);
  const float t  = __builtin_amdgcn_rcpf(fmaf(0.3275911f, ax, 1.0f));
  float p = fmaf(t, 1.061405429f, -1.453152027f);
  p = fmaf(t, p, 1.421413741f);
  p = fmaf(t, p, -0.284496736f);
  p = fmaf(t, p, 0.254829592f);
  p *= t;
  const float e = __expf(-ax * ax);
  const float r = fmaf(-p, e, 1.0f);
  return copysignf(r, x);
}

__device__ __forceinline__ float gelu_exact(float v) {
  return 0.5f * v * (1.0f + erf_fast(v * 0.70710678118654752f));
}

__global__ __launch_bounds__(256)
void pair_mlp_softmax_kernel(const float* __restrict__ x,
                             const float* __restrict__ W1, const float* __restrict__ b1,
                             const float* __restrict__ W2, const float* __restrict__ b2,
                             const float* __restrict__ W3, const float* __restrict__ b3,
                             float* __restrict__ out)
{
  __shared__ __align__(16) _Float16 w1t[D2c * HS];     // W1^T [n][k], 96x96
  __shared__ __align__(16) _Float16 w2t[Dc  * HS];     // W2^T [n][k], 48x96
  __shared__ __align__(16) _Float16 w3t[Dc  * W3S];    // W3^T [n][k], 48x64 (k 48..63 = 0)
  __shared__ __align__(16) float    b1s[D2c];
  __shared__ __align__(16) float    b2s[Dc];
  __shared__ __align__(16) float    b3s[Dc];
  __shared__ __align__(16) _Float16 xish[64];          // x_i as f16 (48 + pad)
  __shared__ __align__(16) _Float16 scratch[8 * 16 * HS]; // per-wave tile scratch

  const int tid = threadIdx.x;
  const int bi  = blockIdx.x;
  const int b   = bi / Nc;
  const int i   = bi % Nc;

  // ---- stage weights (transposed, f16) + biases + x_i into LDS ----
  for (int idx = tid; idx < D2c * D2c; idx += 256) {
    int k = idx / D2c, n = idx % D2c;
    w1t[n * HS + k] = (_Float16)W1[idx];                 // W1[k][n]
  }
  for (int idx = tid; idx < D2c * Dc; idx += 256) {
    int k = idx / Dc, n = idx % Dc;
    w2t[n * HS + k] = (_Float16)W2[idx];                 // W2[k][n]
  }
  for (int idx = tid; idx < Dc * 64; idx += 256) {
    int n = idx % Dc, k = idx / Dc;                      // k in 0..63
    w3t[n * W3S + k] = (k < Dc) ? (_Float16)W3[k * Dc + n] : (_Float16)0.f;
  }
  if (tid < D2c) b1s[tid] = b1[tid];
  if (tid < Dc)  { b2s[tid] = b2[tid]; b3s[tid] = b3[tid]; }
  if (tid < Dc)  xish[tid] = (_Float16)x[((size_t)b * Nc + i) * Dc + tid];
  if (tid >= Dc && tid < 64) xish[tid] = (_Float16)0.f;
  __syncthreads();

  const int wv = tid >> 5;        // wave id 0..7
  const int l  = tid & 31;        // lane
  const int hl = l >> 4;          // half-group (0/1)
  const int nl = l & 15;
  _Float16* scr = scratch + wv * (16 * HS);

  // hoist loop-invariant per-lane biases out of the tile loop
  float bb1[6], bb2[3], bb3[3];
#pragma unroll
  for (int nt = 0; nt < 6; ++nt) bb1[nt] = b1s[nt * 16 + nl];
#pragma unroll
  for (int nt = 0; nt < 3; ++nt) { bb2[nt] = b2s[nt * 16 + nl]; bb3[nt] = b3s[nt * 16 + nl]; }

  const int ntiles = (i >> 4) + 1;   // j-tiles covering j = 0..i
  float m = -1e30f, s = 0.f;
  float ac0 = 0.f, ac1 = 0.f, ac2 = 0.f;   // per-lane weighted accum for d = dt*16 + nl

  for (int t = wv; t < ntiles; t += 8) {
    const int j0 = t << 4;

    // ---- stage x_j tile rows (f16) into scr cols 0..47; lane(hl,nl) does half a row
    {
      const int jr = j0 + nl;
      const int jc = (jr < i) ? jr : i;   // clamp masked rows (weight will be exactly 0)
      const float* xr = x + ((size_t)b * Nc + jc) * Dc + hl * 24;
      _Float16* dst = scr + nl * HS + hl * 24;
#pragma unroll
      for (int q = 0; q < 3; ++q) {
        float4 f0 = ((const float4*)xr)[2 * q];
        float4 f1 = ((const float4*)xr)[2 * q + 1];
        half8 h;
        h[0]=(_Float16)f0.x; h[1]=(_Float16)f0.y; h[2]=(_Float16)f0.z; h[3]=(_Float16)f0.w;
        h[4]=(_Float16)f1.x; h[5]=(_Float16)f1.y; h[6]=(_Float16)f1.z; h[7]=(_Float16)f1.w;
        ((half8*)dst)[q] = h;
      }
    }

    // ---- layer-1 A fragments: pair row = [x_i(0..47) | x_j(0..47)], K-chunks of 32
    const _Float16* xjrow = scr + nl * HS;
    v16h A0 = lds2(xish + hl * 8,          xish + 16 + hl * 8);
    v16h A1 = lds2(xish + 32 + hl * 8,     xjrow + hl * 8);
    v16h A2 = lds2(xjrow + 16 + hl * 8,    xjrow + 32 + hl * 8);

    // ---- layer 1: [16,96]@[96,96]+b1, GELU -> h1 in scr (width 96)
#pragma unroll
    for (int nt = 0; nt < 6; ++nt) {
      v8f c = splat8(bb1[nt]);
      const _Float16* wb = w1t + (nt * 16 + nl) * HS + hl * 8;
      c = wmma_f16(A0, lds2(wb,      wb + 16), c);
      c = wmma_f16(A1, lds2(wb + 32, wb + 48), c);
      c = wmma_f16(A2, lds2(wb + 64, wb + 80), c);
#pragma unroll
      for (int r = 0; r < 8; ++r)
        scr[(r + 8 * hl) * HS + nt * 16 + nl] = (_Float16)gelu_exact(c[r]);
    }

    // ---- layer-2 A fragments from h1 (width 96)
    const _Float16* hrow = scr + nl * HS;
    v16h C0 = lds2(hrow + hl * 8,       hrow + 16 + hl * 8);
    v16h C1 = lds2(hrow + 32 + hl * 8,  hrow + 48 + hl * 8);
    v16h C2 = lds2(hrow + 64 + hl * 8,  hrow + 80 + hl * 8);

    // ---- layer 2: [16,96]@[96,48]+b2, GELU -> h2 in scr (width 64, cols 48..63 zero)
#pragma unroll
    for (int nt = 0; nt < 3; ++nt) {
      v8f c = splat8(bb2[nt]);
      const _Float16* wb = w2t + (nt * 16 + nl) * HS + hl * 8;
      c = wmma_f16(C0, lds2(wb,      wb + 16), c);
      c = wmma_f16(C1, lds2(wb + 32, wb + 48), c);
      c = wmma_f16(C2, lds2(wb + 64, wb + 80), c);
#pragma unroll
      for (int r = 0; r < 8; ++r)
        scr[(r + 8 * hl) * HS + nt * 16 + nl] = (_Float16)gelu_exact(c[r]);
    }
    { // zero K-pad cols 48..63 (one b128 per lane covers all 16 rows x 16 cols)
      half8 z;
#pragma unroll
      for (int e = 0; e < 8; ++e) z[e] = (_Float16)0.f;
      *((half8*)(scr + nl * HS + 48 + hl * 8)) = z;
    }

    // ---- layer 3: [16,64]@[64,48]+b3 -> pair_out tiles P[0..2]
    v16h E0 = lds2(hrow + hl * 8,       hrow + 16 + hl * 8);
    v16h E1 = lds2(hrow + 32 + hl * 8,  hrow + 48 + hl * 8);
    v8f P[3];
#pragma unroll
    for (int nt = 0; nt < 3; ++nt) {
      v8f c = splat8(bb3[nt]);
      const _Float16* wb = w3t + (nt * 16 + nl) * W3S + hl * 8;
      c = wmma_f16(E0, lds2(wb,      wb + 16), c);
      c = wmma_f16(E1, lds2(wb + 32, wb + 48), c);
      P[nt] = c;
    }

    // ---- L2 norm per j (reduce over N-lanes) + online softmax accumulate
    float nr[8];
#pragma unroll
    for (int r = 0; r < 8; ++r) {
      float t2 = P[0][r]*P[0][r] + P[1][r]*P[1][r] + P[2][r]*P[2][r];
      t2 += __shfl_xor(t2, 1);
      t2 += __shfl_xor(t2, 2);
      t2 += __shfl_xor(t2, 4);
      t2 += __shfl_xor(t2, 8);
      const int j = j0 + r + 8 * hl;            // C/D layout: M = r + 8*hl
      nr[r] = (j <= i) ? sqrtf(t2) : -1e30f;
    }
    float tm = nr[0];
#pragma unroll
    for (int r = 1; r < 8; ++r) tm = fmaxf(tm, nr[r]);
    const float mn = fmaxf(m, tm);
    const float sc = __expf(m - mn);
    s *= sc; ac0 *= sc; ac1 *= sc; ac2 *= sc;
#pragma unroll
    for (int r = 0; r < 8; ++r) {
      float p = (nr[r] > -1e29f) ? __expf(nr[r] - mn) : 0.f;
      s += p;
      ac0 += p * P[0][r];
      ac1 += p * P[1][r];
      ac2 += p * P[2][r];
    }
    m = mn;
  }

  // ---- merge half-wave softmax states (lanes L and L+16 hold same d, disjoint j)
  {
    float m2 = __shfl_xor(m, 16);
    float s2 = __shfl_xor(s, 16);
    float d0 = __shfl_xor(ac0, 16);
    float d1 = __shfl_xor(ac1, 16);
    float d2 = __shfl_xor(ac2, 16);
    float mm = fmaxf(m, m2);
    float e1 = __expf(m - mm), e2 = __expf(m2 - mm);
    s   = s   * e1 + s2 * e2;
    ac0 = ac0 * e1 + d0 * e2;
    ac1 = ac1 * e1 + d1 * e2;
    ac2 = ac2 * e1 + d2 * e2;
    m = mm;
  }

  // ---- stash per-wave (m, s, acc[48]) in own scratch, then merge across waves
  float* red = (float*)scr;
  if (l == 0) { red[0] = m; red[1] = s; }
  if (l < 16) {
    red[2 + nl]      = ac0;
    red[2 + 16 + nl] = ac1;
    red[2 + 32 + nl] = ac2;
  }
  __syncthreads();

  if (tid < Dc) {
    float M2 = -1e30f, S = 0.f, A = 0.f;
#pragma unroll
    for (int w = 0; w < 8; ++w) {
      const float* rw = (const float*)(scratch + w * (16 * HS));
      float mw = rw[0], sw = rw[1], aw = rw[2 + tid];
      float Mn = fmaxf(M2, mw);
      float e1 = __expf(M2 - Mn), e2 = __expf(mw - Mn);
      S = S * e1 + sw * e2;
      A = A * e1 + aw * e2;
      M2 = Mn;
    }
    out[((size_t)b * Nc + i) * Dc + tid] = A / S;
  }
}

extern "C" void kernel_launch(void* const* d_in, const int* in_sizes, int n_in,
                              void* d_out, int out_size, void* d_ws, size_t ws_size,
                              hipStream_t stream) {
  (void)in_sizes; (void)n_in; (void)out_size; (void)d_ws; (void)ws_size;
  const float* x  = (const float*)d_in[0];
  const float* W1 = (const float*)d_in[1];
  const float* b1 = (const float*)d_in[2];
  const float* W2 = (const float*)d_in[3];
  const float* b2 = (const float*)d_in[4];
  const float* W3 = (const float*)d_in[5];
  const float* b3 = (const float*)d_in[6];
  float* out = (float*)d_out;
  pair_mlp_softmax_kernel<<<dim3(Bc * Nc), dim3(256), 0, stream>>>(
      x, W1, b1, W2, b2, W3, b3, out);
}
